// GroundedMamba_16999480558031
// MI455X (gfx1250) — matrix operand order repaired and checked by reading, not compiled, not run on hardware
//
#include <hip/hip_runtime.h>
#include <hip/hip_bf16.h>
#include <math.h>

// ---------------------------------------------------------------------------
// GroundedMamba on MI455X (gfx1250).
//  * All projections + LM head: v_wmma_f32_16x16x32_bf16, f32 accumulate.
//    2x4 tile register blocking, ping-pong pipelined, K templated (=512) so
//    the fully-unrolled inner loop uses immediate-offset b128 loads off fixed
//    base addresses (no per-iter address VALU -> no WMMA WAR hazard NOPs).
//  * SSM / WKV recurrences: chunked parallel linear scans (16 chunks x 64
//    steps): ~8x less sequential latency, 16x more wave parallelism.
// ---------------------------------------------------------------------------

typedef __bf16 bf16_t;
typedef __attribute__((ext_vector_type(16))) __bf16 v16bf;
typedef __attribute__((ext_vector_type(8)))  float  v8f;

#define DM_    512
#define SN_    16
#define B_     2
#define L_     1024
#define M_     (B_ * L_)         // 2048 token rows
#define DEPTH_ 8
#define V_     50257
#define VPAD_  50304             // 786 * 64
#define NC_    16                // scan chunks
#define CL_    (L_ / NC_)        // 64 steps per chunk

// ------------------------------ fragments ----------------------------------
union FragU { v16bf v; uint4 q[2]; };

// A operand, row-major [M][K]; two 8-elem contiguous K-runs per lane.
__device__ __forceinline__ v16bf load_a_frag(const bf16_t* rowbase, int k0, int half) {
    FragU u;
    u.q[0] = *reinterpret_cast<const uint4*>(rowbase + k0 + half * 8);
    u.q[1] = *reinterpret_cast<const uint4*>(rowbase + k0 + half * 8 + 16);
    return u.v;
}

// B operand stored N-major (Bt[n][k]); two 8-elem contiguous K-runs per lane.
__device__ __forceinline__ v16bf load_b_frag(const bf16_t* rowbase, int k0, int half) {
    FragU u;
    const uint4* p = reinterpret_cast<const uint4*>(rowbase + k0 + half * 8);
    u.q[0] = p[0];
    u.q[1] = p[2];
    return u.v;
}

// ------------------------------ GEMM ---------------------------------------
// C[M][Nreal] = epilogue(A[M][K] @ Bt[N][K]^T). Wave -> (16*MT) x (16*NT).
// K is compile-time: fully unrolled, immediate-offset loads.
// EPI: 0 store, 1 softplus(x+bias[n]), 2 sigmoid, 3 clip(+-30),
//      4 resid[idx] + scale*x (resid may alias C)
template<int MT, int NT, int EPI, int K>
__global__ __launch_bounds__(128)
void gemm_wmma_kernel(const bf16_t* __restrict__ A, const bf16_t* __restrict__ Bt,
                      float* __restrict__ C, const float* __restrict__ bias,
                      const float* __restrict__ resid, float scale,
                      int M, int Nreal) {
    const int lane = threadIdx.x & 31;
    const int wave = threadIdx.x >> 5;
    const int half = lane >> 4;
    const int l16  = lane & 15;
    const int tm0  = (blockIdx.x * 4 + wave) * MT;   // first 16-row tile
    const int tn0  = blockIdx.y * NT;                // first 16-col tile
    if (tm0 * 16 >= M) return;                       // wave-uniform

    const bf16_t* arow[MT];
#pragma unroll
    for (int i = 0; i < MT; ++i)
        arow[i] = A + (size_t)((tm0 + i) * 16 + l16) * K;
    const bf16_t* brow[NT];
#pragma unroll
    for (int j = 0; j < NT; ++j)
        brow[j] = Bt + (size_t)((tn0 + j) * 16 + l16) * K;

    v8f acc[MT][NT] = {};
    v16bf a0[MT], a1[MT], b0[NT], b1[NT];

#define LOAD_SET(AF, BF, KK)                                         \
    do {                                                             \
        _Pragma("unroll") for (int i = 0; i < MT; ++i)               \
            AF[i] = load_a_frag(arow[i], (KK), half);                \
        _Pragma("unroll") for (int j = 0; j < NT; ++j)               \
            BF[j] = load_b_frag(brow[j], (KK), half);                \
    } while (0)

#define WMMA_SET(AF, BF)                                             \
    do {                                                             \
        _Pragma("unroll") for (int i = 0; i < MT; ++i)               \
            _Pragma("unroll") for (int j = 0; j < NT; ++j)           \
                acc[i][j] = __builtin_amdgcn_wmma_f32_16x16x32_bf16( \
                    false, AF[i], false, BF[j], (short)0,            \
                    acc[i][j], false, false);                        \
    } while (0)

    static_assert(K % 64 == 0, "K must be a multiple of 64");
    LOAD_SET(a0, b0, 0);                       // prologue
#pragma unroll
    for (int kc = 0; kc < K / 64 - 1; ++kc) {  // steady state, compile-time k0
        const int k0 = kc * 64;
        LOAD_SET(a1, b1, k0 + 32);
        WMMA_SET(a0, b0);
        LOAD_SET(a0, b0, k0 + 64);
        WMMA_SET(a1, b1);
    }
    LOAD_SET(a1, b1, K - 32);                  // last chunk pair
    WMMA_SET(a0, b0);
    WMMA_SET(a1, b1);
#undef LOAD_SET
#undef WMMA_SET

#pragma unroll
    for (int i = 0; i < MT; ++i) {
#pragma unroll
        for (int j = 0; j < NT; ++j) {
            int n = (tn0 + j) * 16 + l16;
            if (n >= Nreal) continue;            // vocab pad guard
            float bn = (EPI == 1) ? bias[n] : 0.0f;
#pragma unroll
            for (int v = 0; v < 8; ++v) {
                int m = (tm0 + i) * 16 + half * 8 + v;   // D: M = vgpr + 8*half
                size_t idx = (size_t)m * Nreal + n;
                float val = acc[i][j][v];
                if (EPI == 1) { val += bn; val = (val > 20.f) ? val : log1pf(__expf(val)); }
                else if (EPI == 2) { val = 1.f / (1.f + __expf(-val)); }
                else if (EPI == 3) { val = fminf(30.f, fmaxf(-30.f, val)); }
                else if (EPI == 4) { val = resid[idx] + scale * val; }
                C[idx] = val;
            }
        }
    }
}

// ------------------------------ LayerNorm ----------------------------------
__global__ __launch_bounds__(256)
void ln_kernel(const float* __restrict__ x, const float* __restrict__ g,
               const float* __restrict__ b, float* __restrict__ uf,
               bf16_t* __restrict__ ub) {
    const int row = blockIdx.x;
    const int tid = threadIdx.x;
    const size_t base = (size_t)row * DM_;
    float v0 = x[base + tid];
    float v1 = x[base + tid + 256];
    float s  = v0 + v1;
    float ss = v0 * v0 + v1 * v1;
#pragma unroll
    for (int o = 16; o; o >>= 1) {
        s  += __shfl_xor(s,  o, 32);
        ss += __shfl_xor(ss, o, 32);
    }
    __shared__ float sh_s[8], sh_ss[8];
    if ((tid & 31) == 0) { sh_s[tid >> 5] = s; sh_ss[tid >> 5] = ss; }
    __syncthreads();
    s = 0.f; ss = 0.f;
#pragma unroll
    for (int i = 0; i < 8; ++i) { s += sh_s[i]; ss += sh_ss[i]; }
    float mu  = s * (1.0f / DM_);
    float var = ss * (1.0f / DM_) - mu * mu;
    float rs  = rsqrtf(var + 1e-5f);
    float o0 = (v0 - mu) * rs * g[tid]       + b[tid];
    float o1 = (v1 - mu) * rs * g[tid + 256] + b[tid + 256];
    if (uf) { uf[base + tid] = o0; uf[base + tid + 256] = o1; }
    ub[base + tid]       = (bf16_t)o0;
    ub[base + tid + 256] = (bf16_t)o1;
}

// --------------------- SSM chunked parallel scan ---------------------------
// h_t = g_t*h_{t-1} + c_t, g_t = exp(dt_t*A[d][n]), c_t = dt_t*u_t*B[t][n]
__global__ __launch_bounds__(256)
void ssm_scan_p1(const float* __restrict__ u,  const float* __restrict__ dt,
                 const float* __restrict__ Bm, const float* __restrict__ A_log,
                 float* __restrict__ Gq, float* __restrict__ Hend) {
    int gid = blockIdx.x * 256 + threadIdx.x;
    if (gid >= B_ * DM_ * SN_ * NC_) return;
    int n = gid & 15;
    int rest = gid >> 4;
    int c  = rest % NC_;
    int ch = rest / NC_;
    int b = ch / DM_, d = ch % DM_;
    float A = -__expf(A_log[d * SN_ + n]);
    float G = 1.f, h = 0.f;
    const size_t baseU = ((size_t)b * L_ + c * CL_) * DM_ + d;
    const size_t baseN = ((size_t)b * L_ + c * CL_) * SN_ + n;
    for (int tt = 0; tt < CL_; ++tt) {
        float dt_t = dt[baseU + (size_t)tt * DM_];
        float u_t  = u [baseU + (size_t)tt * DM_];
        float Bt   = Bm[baseN + (size_t)tt * SN_];
        float g = __expf(dt_t * A);
        h = g * h + (dt_t * u_t) * Bt;
        G *= g;
    }
    Gq[gid] = G;
    Hend[gid] = h;
}

__global__ __launch_bounds__(256)
void ssm_scan_p2(const float* __restrict__ Gq, const float* __restrict__ Hend,
                 float* __restrict__ Hin) {
    int gid = blockIdx.x * 256 + threadIdx.x;
    if (gid >= B_ * DM_ * SN_) return;
    int n = gid & 15;
    int ch = gid >> 4;
    size_t base = ((size_t)ch * NC_) * 16 + n;
    float h = 0.f;
    for (int c = 0; c < NC_; ++c) {
        size_t s = base + (size_t)c * 16;
        Hin[s] = h;
        h = Gq[s] * h + Hend[s];
    }
}

__global__ __launch_bounds__(256)
void ssm_scan_p3(const float* __restrict__ u,  const float* __restrict__ dt,
                 const float* __restrict__ Bm, const float* __restrict__ Cm,
                 const float* __restrict__ A_log, const float* __restrict__ Dp,
                 const float* __restrict__ Hin, float* __restrict__ x) {
    int gid = blockIdx.x * 256 + threadIdx.x;
    if (gid >= B_ * DM_ * SN_ * NC_) return;
    int n = gid & 15;
    int rest = gid >> 4;
    int c  = rest % NC_;
    int ch = rest / NC_;
    int b = ch / DM_, d = ch % DM_;
    float A  = -__expf(A_log[d * SN_ + n]);
    float Dv = Dp[d];
    float h  = Hin[gid];
    const size_t baseU = ((size_t)b * L_ + c * CL_) * DM_ + d;
    const size_t baseN = ((size_t)b * L_ + c * CL_) * SN_ + n;
    for (int tt = 0; tt < CL_; ++tt) {
        float dt_t = dt[baseU + (size_t)tt * DM_];
        float u_t  = u [baseU + (size_t)tt * DM_];
        float Bt   = Bm[baseN + (size_t)tt * SN_];
        float Ct   = Cm[baseN + (size_t)tt * SN_];
        h = __expf(dt_t * A) * h + (dt_t * u_t) * Bt;
        float y = h * Ct;
        y += __shfl_xor(y, 8, 16);
        y += __shfl_xor(y, 4, 16);
        y += __shfl_xor(y, 2, 16);
        y += __shfl_xor(y, 1, 16);
        if (n == 0) x[baseU + (size_t)tt * DM_] += 0.5f * (y + Dv * u_t);
    }
}

// --------------------- WKV chunked parallel scan ---------------------------
// a_t = w*a_{t-1} + ek_t*v_t ; b_t = w*b_{t-1} + ek_t  (w const per channel)
__global__ __launch_bounds__(256)
void wkv_scan_p1(const float* __restrict__ k, const float* __restrict__ v,
                 const float* __restrict__ t_decay,
                 float* __restrict__ Aend, float* __restrict__ Bend) {
    int gid = blockIdx.x * 256 + threadIdx.x;
    if (gid >= B_ * DM_ * NC_) return;
    int d = gid % DM_;
    int rest = gid / DM_;
    int c = rest % NC_;
    int b = rest / NC_;
    float w = __expf(-__expf(t_decay[d]));
    float a = 0.f, bb = 0.f;
    const size_t base = ((size_t)b * L_ + c * CL_) * DM_ + d;
    for (int tt = 0; tt < CL_; ++tt) {
        size_t i = base + (size_t)tt * DM_;
        float ek = __expf(k[i]);
        a  = w * a  + ek * v[i];
        bb = w * bb + ek;
    }
    Aend[gid] = a;
    Bend[gid] = bb;
}

__global__ __launch_bounds__(256)
void wkv_scan_p2(const float* __restrict__ t_decay,
                 const float* __restrict__ Aend, const float* __restrict__ Bend,
                 float* __restrict__ Ain, float* __restrict__ Bin) {
    int gid = blockIdx.x * 256 + threadIdx.x;
    if (gid >= B_ * DM_) return;
    int d = gid % DM_;
    int b = gid / DM_;
    float wc = __expf(-(float)CL_ * __expf(t_decay[d]));   // w^CL
    float a = 0.f, bb = 0.f;
    for (int c = 0; c < NC_; ++c) {
        size_t s = ((size_t)b * NC_ + c) * DM_ + d;
        Ain[s] = a;
        Bin[s] = bb;
        a  = wc * a  + Aend[s];
        bb = wc * bb + Bend[s];
    }
}

__global__ __launch_bounds__(256)
void wkv_scan_p3(const float* __restrict__ r, const float* __restrict__ k,
                 const float* __restrict__ v, const float* __restrict__ t_decay,
                 const float* __restrict__ t_first,
                 const float* __restrict__ Ain, const float* __restrict__ Bin,
                 bf16_t* __restrict__ ro_bf) {
    int gid = blockIdx.x * 256 + threadIdx.x;
    if (gid >= B_ * DM_ * NC_) return;
    int d = gid % DM_;
    int rest = gid / DM_;
    int c = rest % NC_;
    int b = rest / NC_;
    float w  = __expf(-__expf(t_decay[d]));
    float tf = t_first[d];
    float a = Ain[gid], bb = Bin[gid];
    const size_t base = ((size_t)b * L_ + c * CL_) * DM_ + d;
    for (int tt = 0; tt < CL_; ++tt) {
        size_t i = base + (size_t)tt * DM_;
        float kt = k[i], vt = v[i];
        float ek = __expf(kt);
        float eu = __expf(tf + kt);
        float out = (a + eu * vt) / (bb + eu + 1e-6f);
        a  = w * a  + ek * vt;
        bb = w * bb + ek;
        ro_bf[i] = (bf16_t)(r[i] * out);
    }
}

// ------------------------ conversions / gather -----------------------------
__global__ __launch_bounds__(256)
void embed_kernel(const int* __restrict__ ids, const float* __restrict__ emb,
                  float* __restrict__ x) {
    size_t idx = (size_t)blockIdx.x * blockDim.x + threadIdx.x;
    if (idx >= (size_t)M_ * DM_) return;
    size_t m = idx / DM_, d = idx % DM_;
    x[idx] = emb[(size_t)ids[m] * DM_ + d];
}

// f32 W[layer][k][n] -> bf16 Wt[layer][n][k]  (N-major for contiguous B frags)
__global__ __launch_bounds__(256)
void wconv_t_kernel(const float* __restrict__ W, bf16_t* __restrict__ Wt,
                    int K, int N, int layers) {
    size_t idx = (size_t)blockIdx.x * blockDim.x + threadIdx.x;
    size_t lsz = (size_t)K * N;
    if (idx >= lsz * layers) return;
    size_t l = idx / lsz, rem = idx % lsz;
    size_t k = rem / N, n = rem % N;
    Wt[l * lsz + n * (size_t)K + k] = (bf16_t)W[idx];
}

__global__ __launch_bounds__(256)
void embconv_kernel(const float* __restrict__ emb, bf16_t* __restrict__ embbf) {
    size_t idx = (size_t)blockIdx.x * blockDim.x + threadIdx.x;
    if (idx >= (size_t)VPAD_ * DM_) return;
    size_t row = idx / DM_, d = idx % DM_;
    embbf[idx] = (row < V_) ? (bf16_t)emb[row * (size_t)DM_ + d] : (bf16_t)0.0f;
}

// ------------------------------ host driver --------------------------------
extern "C" void kernel_launch(void* const* d_in, const int* in_sizes, int n_in,
                              void* d_out, int out_size, void* d_ws, size_t ws_size,
                              hipStream_t stream) {
    const int*   ids     = (const int*)  d_in[0];
    const float* emb     = (const float*)d_in[1];
    const float* ln1_g   = (const float*)d_in[2];
    const float* ln1_b   = (const float*)d_in[3];
    const float* W_dt    = (const float*)d_in[4];
    const float* b_dt    = (const float*)d_in[5];
    const float* W_B     = (const float*)d_in[6];
    const float* W_C     = (const float*)d_in[7];
    const float* A_log   = (const float*)d_in[8];
    const float* D_p     = (const float*)d_in[9];
    const float* ln2_g   = (const float*)d_in[10];
    const float* ln2_b   = (const float*)d_in[11];
    const float* W_r     = (const float*)d_in[12];
    const float* W_k     = (const float*)d_in[13];
    const float* W_v     = (const float*)d_in[14];
    const float* W_o     = (const float*)d_in[15];
    const float* t_decay = (const float*)d_in[16];
    const float* t_first = (const float*)d_in[17];
    const float* lnf_g   = (const float*)d_in[18];
    const float* lnf_b   = (const float*)d_in[19];
    float* out = (float*)d_out;

    // ---- workspace carving (256B aligned) ----
    char* ws = (char*)d_ws;
    size_t off = 0;
    auto carve = [&](size_t bytes) -> char* {
        off = (off + 255) & ~(size_t)255;
        char* p = ws + off;
        off += bytes;
        return p;
    };
    float*  x    = (float*) carve((size_t)M_ * DM_ * 4);
    float*  u    = (float*) carve((size_t)M_ * DM_ * 4);
    bf16_t* ubf  = (bf16_t*)carve((size_t)M_ * DM_ * 2);
    float*  dtb  = (float*) carve((size_t)M_ * DM_ * 4);
    float*  Bm   = (float*) carve((size_t)M_ * SN_ * 4);
    float*  Cm   = (float*) carve((size_t)M_ * SN_ * 4);
    float*  rb   = (float*) carve((size_t)M_ * DM_ * 4);
    float*  kb   = (float*) carve((size_t)M_ * DM_ * 4);
    float*  vb   = (float*) carve((size_t)M_ * DM_ * 4);
    bf16_t* robf = (bf16_t*)carve((size_t)M_ * DM_ * 2);
    float*  Gq   = (float*) carve((size_t)B_ * DM_ * SN_ * NC_ * 4);
    float*  Hend = (float*) carve((size_t)B_ * DM_ * SN_ * NC_ * 4);
    float*  Hin  = (float*) carve((size_t)B_ * DM_ * SN_ * NC_ * 4);
    float*  Aend = (float*) carve((size_t)B_ * DM_ * NC_ * 4);
    float*  Bend = (float*) carve((size_t)B_ * DM_ * NC_ * 4);
    float*  Ain  = (float*) carve((size_t)B_ * DM_ * NC_ * 4);
    float*  Bin  = (float*) carve((size_t)B_ * DM_ * NC_ * 4);
    bf16_t* WtDt = (bf16_t*)carve((size_t)DEPTH_ * DM_ * DM_ * 2);
    bf16_t* WtB  = (bf16_t*)carve((size_t)DEPTH_ * DM_ * SN_ * 2);
    bf16_t* WtC  = (bf16_t*)carve((size_t)DEPTH_ * DM_ * SN_ * 2);
    bf16_t* Wtr  = (bf16_t*)carve((size_t)DEPTH_ * DM_ * DM_ * 2);
    bf16_t* Wtk  = (bf16_t*)carve((size_t)DEPTH_ * DM_ * DM_ * 2);
    bf16_t* Wtv  = (bf16_t*)carve((size_t)DEPTH_ * DM_ * DM_ * 2);
    bf16_t* Wto  = (bf16_t*)carve((size_t)DEPTH_ * DM_ * DM_ * 2);
    bf16_t* embbf= (bf16_t*)carve((size_t)VPAD_ * DM_ * 2);
    (void)ws_size; (void)n_in; (void)in_sizes; (void)out_size;

    // ---- one-shot weight conversion (bf16, N-major) ----
    {
        size_t big = (size_t)DEPTH_ * DM_ * DM_;
        int gb = (int)((big + 255) / 256);
        wconv_t_kernel<<<gb, 256, 0, stream>>>(W_dt, WtDt, DM_, DM_, DEPTH_);
        wconv_t_kernel<<<gb, 256, 0, stream>>>(W_r,  Wtr,  DM_, DM_, DEPTH_);
        wconv_t_kernel<<<gb, 256, 0, stream>>>(W_k,  Wtk,  DM_, DM_, DEPTH_);
        wconv_t_kernel<<<gb, 256, 0, stream>>>(W_v,  Wtv,  DM_, DM_, DEPTH_);
        wconv_t_kernel<<<gb, 256, 0, stream>>>(W_o,  Wto,  DM_, DM_, DEPTH_);
        size_t sml = (size_t)DEPTH_ * DM_ * SN_;
        int gs = (int)((sml + 255) / 256);
        wconv_t_kernel<<<gs, 256, 0, stream>>>(W_B, WtB, DM_, SN_, DEPTH_);
        wconv_t_kernel<<<gs, 256, 0, stream>>>(W_C, WtC, DM_, SN_, DEPTH_);
        size_t et = (size_t)VPAD_ * DM_;
        embconv_kernel<<<(int)((et + 255) / 256), 256, 0, stream>>>(emb, embbf);
    }

    // ---- embedding gather ----
    embed_kernel<<<(M_ * DM_) / 256, 256, 0, stream>>>(ids, emb, x);

    const dim3 blk128(128);
    const dim3 gFull(M_ / 128, DM_ / 64);   // MT=2,NT=4: 32x64 per wave
    const dim3 gSmall(M_ / 128, 1);         // MT=2,NT=1: N=16 GEMMs
    const int  gScan1 = (B_ * DM_ * SN_ * NC_) / 256;   // 1024 blocks
    const int  gScan2 = (B_ * DM_ * SN_) / 256;         // 64 blocks
    const int  gWkv13 = (B_ * DM_ * NC_) / 256;         // 64 blocks
    const int  gWkv2  = (B_ * DM_) / 256;               // 4 blocks

    for (int i = 0; i < DEPTH_; ++i) {
        const size_t wBig = (size_t)i * DM_ * DM_;
        const size_t wSml = (size_t)i * DM_ * SN_;
        // --- SSM branch ---
        ln_kernel<<<M_, 256, 0, stream>>>(x, ln1_g + i * DM_, ln1_b + i * DM_, u, ubf);
        gemm_wmma_kernel<2, 4, 1, DM_><<<gFull, blk128, 0, stream>>>(
            ubf, WtDt + wBig, dtb, b_dt + i * DM_, nullptr, 0.f, M_, DM_);
        gemm_wmma_kernel<2, 1, 0, DM_><<<gSmall, blk128, 0, stream>>>(
            ubf, WtB + wSml, Bm, nullptr, nullptr, 0.f, M_, SN_);
        gemm_wmma_kernel<2, 1, 0, DM_><<<gSmall, blk128, 0, stream>>>(
            ubf, WtC + wSml, Cm, nullptr, nullptr, 0.f, M_, SN_);
        ssm_scan_p1<<<gScan1, 256, 0, stream>>>(u, dtb, Bm,
            A_log + (size_t)i * DM_ * SN_, Gq, Hend);
        ssm_scan_p2<<<gScan2, 256, 0, stream>>>(Gq, Hend, Hin);
        ssm_scan_p3<<<gScan1, 256, 0, stream>>>(u, dtb, Bm, Cm,
            A_log + (size_t)i * DM_ * SN_, D_p + i * DM_, Hin, x);
        // --- WKV branch ---
        ln_kernel<<<M_, 256, 0, stream>>>(x, ln2_g + i * DM_, ln2_b + i * DM_, nullptr, ubf);
        gemm_wmma_kernel<2, 4, 2, DM_><<<gFull, blk128, 0, stream>>>(
            ubf, Wtr + wBig, rb, nullptr, nullptr, 0.f, M_, DM_);
        gemm_wmma_kernel<2, 4, 3, DM_><<<gFull, blk128, 0, stream>>>(
            ubf, Wtk + wBig, kb, nullptr, nullptr, 0.f, M_, DM_);
        gemm_wmma_kernel<2, 4, 0, DM_><<<gFull, blk128, 0, stream>>>(
            ubf, Wtv + wBig, vb, nullptr, nullptr, 0.f, M_, DM_);
        wkv_scan_p1<<<gWkv13, 256, 0, stream>>>(kb, vb, t_decay + i * DM_, Aend, Bend);
        wkv_scan_p2<<<gWkv2, 256, 0, stream>>>(t_decay + i * DM_, Aend, Bend, Ain, Bin);
        wkv_scan_p3<<<gWkv13, 256, 0, stream>>>(rb, kb, vb, t_decay + i * DM_,
            t_first + i * DM_, Ain, Bin, robf);
        gemm_wmma_kernel<2, 4, 4, DM_><<<gFull, blk128, 0, stream>>>(
            robf, Wto + wBig, x, nullptr, x, 0.5f, M_, DM_);
    }

    // ---- final LN + tied LM head over padded vocab ----
    ln_kernel<<<M_, 256, 0, stream>>>(x, lnf_g, lnf_b, nullptr, ubf);
    dim3 gHead(M_ / 128, VPAD_ / 64);
    gemm_wmma_kernel<2, 4, 0, DM_><<<gHead, blk128, 0, stream>>>(
        ubf, embbf, out, nullptr, nullptr, 0.f, M_, V_);
}
